// GraphEmbedding_2525440770664
// MI455X (gfx1250) — matrix operand order repaired and checked by reading, not compiled, not run on hardware
//
#include <hip/hip_runtime.h>
#include <hip/hip_bf16.h>
#include <math.h>

typedef _Float16 v16h __attribute__((ext_vector_type(16)));
typedef _Float16 h8   __attribute__((ext_vector_type(8)));
typedef _Float16 h4   __attribute__((ext_vector_type(4)));
typedef float    v8f  __attribute__((ext_vector_type(8)));
typedef float    f4   __attribute__((ext_vector_type(4)));

#define TGN_B 4096
#define TGN_K 20
#define TGN_D 128
#define NGH_ROWS (TGN_B * TGN_K)   // 81920

union AB16 { v16h v; h8 h[2]; };

// ---------------------------------------------------------------------------
// CDNA5 async global->LDS copy (16B per lane), tracked by ASYNCcnt.
// ---------------------------------------------------------------------------
__device__ __forceinline__ void lds_async_b128(void* lds_dst, const void* gsrc) {
  unsigned l = (unsigned)(unsigned long long)lds_dst;   // LDS byte address
  asm volatile("global_load_async_to_lds_b128 %0, %1, off"
               :: "v"(l), "v"(gsrc) : "memory");
}
__device__ __forceinline__ void wait_async_zero() {
  asm volatile("s_wait_asynccnt 0" ::: "memory");
}

// ---------------------------------------------------------------------------
// Core: one wave computes a 16x16 f32 tile of  A[16,Kdim] @ W[Kdim,128] for
// output columns [n0, n0+16).  A is row-major f16 in LDS, W is transposed
// f16 [128][Kdim] in LDS.  Fragment layouts per CDNA5 ISA 7.12.2:
//   A lane L: M=L&15, halves e: K = e + 8*(e>>3) + 8*(L>>4)   (+kc)
//   B lane L: N=n0+(L&15), halves e: K = e + 8*(e>>3) + 8*(L>>4)   (+kc)
// ---------------------------------------------------------------------------
__device__ __forceinline__ v8f wmma_tile(const _Float16* __restrict__ As, int lda,
                                         const _Float16* __restrict__ WT, int Kdim,
                                         int n0, v8f acc) {
  const int lane = threadIdx.x & 31;
  const int hi   = lane >> 4;
  const int row  = lane & 15;
  const int col  = n0 + row;
  const _Float16* arow = As + row * lda + 8 * hi;
  const _Float16* bcol = WT + col * Kdim + 8 * hi;
#pragma unroll 4
  for (int kc = 0; kc < Kdim; kc += 32) {
    AB16 a, b;
    a.h[0] = *(const h8*)(arow + kc);
    a.h[1] = *(const h8*)(arow + kc + 16);
    b.h[0] = *(const h8*)(bcol + kc);
    b.h[1] = *(const h8*)(bcol + kc + 16);
    acc = __builtin_amdgcn_wmma_f32_16x16x32_f16(false, a.v, false, b.v,
                                                 (short)0, acc, false, false);
  }
  return acc;
}

// D tile layout: VGPR r -> M = r + 8*(lane>>4), N = n0 + (lane&15)
__device__ __forceinline__ void tile_to_lds_f16(v8f acc, _Float16* dst, int ld, int n0,
                                                const float* bias, bool relu) {
  const int lane = threadIdx.x & 31;
  const int m0   = (lane >> 4) * 8;
  const int col  = n0 + (lane & 15);
  float bv = bias ? bias[col] : 0.0f;
#pragma unroll
  for (int r = 0; r < 8; ++r) {
    float x = acc[r] + bv;
    if (relu) x = fmaxf(x, 0.0f);
    dst[(m0 + r) * ld + col] = (_Float16)x;
  }
}

__device__ __forceinline__ void tile_to_glb_f16(v8f acc, _Float16* dst, int n0) {
  const int lane = threadIdx.x & 31;
  const int m0   = (lane >> 4) * 8;
  const int col  = n0 + (lane & 15);
#pragma unroll
  for (int r = 0; r < 8; ++r)
    dst[(size_t)(m0 + r) * 128 + col] = (_Float16)acc[r];
}

__device__ __forceinline__ void tile_to_glb_f32(v8f acc, float* dst, int n0,
                                                const float* bias) {
  const int lane = threadIdx.x & 31;
  const int m0   = (lane >> 4) * 8;
  const int col  = n0 + (lane & 15);
  float bv = bias ? bias[col] : 0.0f;
#pragma unroll
  for (int r = 0; r < 8; ++r)
    dst[(size_t)(m0 + r) * 128 + col] = acc[r] + bv;
}

// ---------------------------------------------------------------------------
// Weight conversion: f32 [K][N] row-major -> f16 transposed [N][K]
// ---------------------------------------------------------------------------
__global__ void convert_wT(const float* __restrict__ w, _Float16* __restrict__ wT,
                           int K, int N) {
  int idx = blockIdx.x * blockDim.x + threadIdx.x;
  if (idx < K * N) {
    int k = idx / N, n = idx % N;
    wT[n * K + k] = (_Float16)w[idx];
  }
}

// ---------------------------------------------------------------------------
// Source pipeline: src_emb = MLP([nf||mem]) ; q = [src_emb||cos(dt*w+b)] @ wq
// ---------------------------------------------------------------------------
__global__ void src_pipeline(const float* __restrict__ nf, const float* __restrict__ mem,
                             const int* __restrict__ src,
                             const float* __restrict__ ts, const float* __restrict__ lu,
                             const float* __restrict__ wts, const float* __restrict__ bts,
                             const _Float16* __restrict__ w1Tg,
                             const _Float16* __restrict__ w2Tg,
                             const _Float16* __restrict__ wqTg,
                             const float* __restrict__ b1, const float* __restrict__ b2,
                             float* __restrict__ src_emb, float* __restrict__ qout,
                             int blocks_per_wg) {
  extern __shared__ _Float16 sm[];
  _Float16* w1T = sm;                 // 256*128
  _Float16* w2T = w1T + 256 * 128;    // 128*128
  _Float16* wqT = w2T + 128 * 128;    // 256*128
  _Float16* A1  = wqT + 256 * 128;    // 16*256
  _Float16* H   = A1 + 16 * 256;      // 16*128
  _Float16* Aq  = H + 16 * 128;       // 16*256
  const int t = threadIdx.x, w = t >> 5;
  // async weight staging: 16B per lane per issue, <=40 outstanding per wave
  for (int i = t; i < 4096; i += 256) {
    lds_async_b128(w1T + i * 8, w1Tg + i * 8);
    lds_async_b128(wqT + i * 8, wqTg + i * 8);
  }
  for (int i = t; i < 2048; i += 256)
    lds_async_b128(w2T + i * 8, w2Tg + i * 8);
  wait_async_zero();
  __syncthreads();
  for (int blk = 0; blk < blocks_per_wg; ++blk) {
    const int r0 = (blockIdx.x * blocks_per_wg + blk) * 16;
    if (r0 >= TGN_B) break;
    // gather [nf || mem] (float4 -> h4) and time-encode into Aq[:,128:256]
    for (int i = t; i < 16 * 64; i += 256) {
      int row = i >> 6, c4 = (i & 63) * 4;
      int node = src[r0 + row];
      f4 v = (c4 < 128) ? *(const f4*)(nf + (size_t)node * 128 + c4)
                        : *(const f4*)(mem + (size_t)node * 128 + (c4 - 128));
      *(h4*)(A1 + row * 256 + c4) = __builtin_convertvector(v, h4);
    }
    for (int i = t; i < 16 * 128; i += 256) {
      int row = i >> 7, j = i & 127;
      float dt = ts[r0 + row] - lu[src[r0 + row]];
      Aq[row * 256 + 128 + j] = (_Float16)cosf(dt * wts[j] + bts[j]);
    }
    __syncthreads();
    v8f acc = {};
    acc = wmma_tile(A1, 256, w1T, 256, w * 16, acc);
    tile_to_lds_f16(acc, H, 128, w * 16, b1, true);
    __syncthreads();
    v8f acc2 = {};
    acc2 = wmma_tile(H, 128, w2T, 128, w * 16, acc2);
    {   // src_emb: f32 to ws AND f16 into Aq[:,0:128]
      const int lane = t & 31, m0 = (lane >> 4) * 8, col = w * 16 + (lane & 15);
      float bv = b2[col];
#pragma unroll
      for (int r = 0; r < 8; ++r) {
        float x = acc2[r] + bv;
        src_emb[(size_t)(r0 + m0 + r) * 128 + col] = x;
        Aq[(m0 + r) * 256 + col] = (_Float16)x;
      }
    }
    __syncthreads();
    v8f acc3 = {};
    acc3 = wmma_tile(Aq, 256, wqT, 256, w * 16, acc3);
    tile_to_glb_f32(acc3, qout + (size_t)r0 * 128, w * 16, nullptr);
    __syncthreads();
  }
}

// ---------------------------------------------------------------------------
// Neighbor merge MLP: emb = relu([nf||mem] @ w1 + b1) @ w2 + b2   (f16 out)
// ---------------------------------------------------------------------------
__global__ void ngh_mlp(const float* __restrict__ nf, const float* __restrict__ mem,
                        const int* __restrict__ ngh,
                        const _Float16* __restrict__ w1Tg,
                        const _Float16* __restrict__ w2Tg,
                        const float* __restrict__ b1, const float* __restrict__ b2,
                        _Float16* __restrict__ emb_out, int blocks_per_wg) {
  extern __shared__ _Float16 sm[];
  _Float16* w1T = sm;                // 256*128
  _Float16* w2T = w1T + 256 * 128;   // 128*128
  _Float16* A1  = w2T + 128 * 128;   // 16*256
  _Float16* H   = A1 + 16 * 256;     // 16*128
  const int t = threadIdx.x, w = t >> 5;
  for (int i = t; i < 4096; i += 256)
    lds_async_b128(w1T + i * 8, w1Tg + i * 8);
  for (int i = t; i < 2048; i += 256)
    lds_async_b128(w2T + i * 8, w2Tg + i * 8);
  wait_async_zero();
  __syncthreads();
  for (int blk = 0; blk < blocks_per_wg; ++blk) {
    const int r0 = (blockIdx.x * blocks_per_wg + blk) * 16;
    if (r0 >= NGH_ROWS) break;
    for (int i = t; i < 16 * 64; i += 256) {
      int row = i >> 6, c4 = (i & 63) * 4;
      int node = ngh[r0 + row];
      f4 v = (c4 < 128) ? *(const f4*)(nf + (size_t)node * 128 + c4)
                        : *(const f4*)(mem + (size_t)node * 128 + (c4 - 128));
      *(h4*)(A1 + row * 256 + c4) = __builtin_convertvector(v, h4);
    }
    __syncthreads();
    v8f acc = {};
    acc = wmma_tile(A1, 256, w1T, 256, w * 16, acc);
    tile_to_lds_f16(acc, H, 128, w * 16, b1, true);
    __syncthreads();
    v8f acc2 = {};
    acc2 = wmma_tile(H, 128, w2T, 128, w * 16, acc2);
    {
      const int lane = t & 31, m0 = (lane >> 4) * 8, col = w * 16 + (lane & 15);
      float bv = b2[col];
#pragma unroll
      for (int r = 0; r < 8; ++r)
        emb_out[(size_t)(r0 + m0 + r) * 128 + col] = (_Float16)(acc2[r] + bv);
    }
    __syncthreads();
  }
}

// ---------------------------------------------------------------------------
// k/v projection: keyv = [emb || cos(dt*w+b) || edge_f] (K=384) @ wk / wv
// ---------------------------------------------------------------------------
__global__ void kv_proj(const _Float16* __restrict__ emb,
                        const float* __restrict__ ts, const float* __restrict__ et,
                        const float* __restrict__ ef, const int* __restrict__ eidx,
                        const float* __restrict__ wtn, const float* __restrict__ btn,
                        const _Float16* __restrict__ wkTg,
                        const _Float16* __restrict__ wvTg,
                        _Float16* __restrict__ kout, _Float16* __restrict__ vout,
                        int blocks_per_wg) {
  extern __shared__ _Float16 sm[];
  _Float16* wkT = sm;                // 384*128
  _Float16* wvT = wkT + 384 * 128;   // 384*128
  _Float16* A   = wvT + 384 * 128;   // 16*384
  const int t = threadIdx.x, w = t >> 5;
  for (int i = t; i < 6144; i += 256) {   // 24 issues/lane per matrix
    lds_async_b128(wkT + i * 8, wkTg + i * 8);
    lds_async_b128(wvT + i * 8, wvTg + i * 8);
  }
  wait_async_zero();
  __syncthreads();
  const int lane = t & 31;
  const int hi   = lane >> 4;
  const int row  = lane & 15;
  const int col  = w * 16 + row;
  for (int blk = 0; blk < blocks_per_wg; ++blk) {
    const int r0 = (blockIdx.x * blocks_per_wg + blk) * 16;
    if (r0 >= NGH_ROWS) break;
    // emb segment: h8 copies (one pass, 256 chunks)
    for (int i = t; i < 16 * 16; i += 256) {
      int rr = i >> 4, c8 = (i & 15) * 8;
      *(h8*)(A + rr * 384 + c8) = *(const h8*)(emb + (size_t)(r0 + rr) * 128 + c8);
    }
    // time-encode segment
    for (int i = t; i < 16 * 128; i += 256) {
      int rr = i >> 7, j = i & 127;
      int fr = r0 + rr;
      float dt = ts[fr / TGN_K] - et[fr];
      A[rr * 384 + 128 + j] = (_Float16)cosf(dt * wtn[j] + btn[j]);
    }
    // edge feature segment: float4 -> h4
    for (int i = t; i < 16 * 32; i += 256) {
      int rr = i >> 5, c4 = (i & 31) * 4;
      f4 v = *(const f4*)(ef + (size_t)eidx[r0 + rr] * 128 + c4);
      *(h4*)(A + rr * 384 + 256 + c4) = __builtin_convertvector(v, h4);
    }
    __syncthreads();
    v8f ak = {}, av = {};
    const _Float16* arow = A + row * 384 + 8 * hi;
    const _Float16* bk   = wkT + col * 384 + 8 * hi;
    const _Float16* bv   = wvT + col * 384 + 8 * hi;
#pragma unroll 4
    for (int kc = 0; kc < 384; kc += 32) {   // shared A fragment, two WMMAs
      AB16 a, b0, b1;
      a.h[0]  = *(const h8*)(arow + kc);
      a.h[1]  = *(const h8*)(arow + kc + 16);
      b0.h[0] = *(const h8*)(bk + kc);
      b0.h[1] = *(const h8*)(bk + kc + 16);
      b1.h[0] = *(const h8*)(bv + kc);
      b1.h[1] = *(const h8*)(bv + kc + 16);
      ak = __builtin_amdgcn_wmma_f32_16x16x32_f16(false, a.v, false, b0.v,
                                                  (short)0, ak, false, false);
      av = __builtin_amdgcn_wmma_f32_16x16x32_f16(false, a.v, false, b1.v,
                                                  (short)0, av, false, false);
    }
    tile_to_glb_f16(ak, kout + (size_t)r0 * 128, w * 16);
    tile_to_glb_f16(av, vout + (size_t)r0 * 128, w * 16);
    __syncthreads();
  }
}

// ---------------------------------------------------------------------------
// Attention: one wave per batch row, H=2 heads of 64 dims (lanes 0-15 / 16-31)
// ---------------------------------------------------------------------------
__global__ void attn_kernel(const float* __restrict__ qg,
                            const _Float16* __restrict__ kg,
                            const _Float16* __restrict__ vg,
                            const int* __restrict__ neighbors,
                            _Float16* __restrict__ ctx_out) {
  const int wid  = blockIdx.x * (blockDim.x >> 5) + (threadIdx.x >> 5);
  const int lane = threadIdx.x & 31;
  if (wid >= TGN_B) return;
  const int b = wid;
  f4 q0 = *(const f4*)(qg + (size_t)b * 128 + lane * 4);
  float scores[TGN_K];
#pragma unroll 4
  for (int n = 0; n < TGN_K; ++n) {
    h4 k4 = *(const h4*)(kg + ((size_t)b * TGN_K + n) * 128 + lane * 4);
    float p = q0[0] * (float)k4[0] + q0[1] * (float)k4[1] +
              q0[2] * (float)k4[2] + q0[3] * (float)k4[3];
    p += __shfl_xor(p, 1, 32);
    p += __shfl_xor(p, 2, 32);
    p += __shfl_xor(p, 4, 32);
    p += __shfl_xor(p, 8, 32);     // reduce within each 16-lane head group
    float s = p * 0.125f;          // 1/sqrt(64)
    if (neighbors[b * TGN_K + n] == 0) s = -1e9f;
    scores[n] = s;
  }
  float m = -1e30f;
#pragma unroll
  for (int n = 0; n < TGN_K; ++n) m = fmaxf(m, scores[n]);
  float denom = 0.0f, e[TGN_K];
#pragma unroll
  for (int n = 0; n < TGN_K; ++n) { e[n] = __expf(scores[n] - m); denom += e[n]; }
  const float inv = (m < -5e8f) ? 0.0f : (1.0f / denom);   // all-masked -> zeros
  float c0 = 0.f, c1 = 0.f, c2 = 0.f, c3 = 0.f;
#pragma unroll 4
  for (int n = 0; n < TGN_K; ++n) {
    const float a = e[n] * inv;
    h4 v4 = *(const h4*)(vg + ((size_t)b * TGN_K + n) * 128 + lane * 4);
    c0 += a * (float)v4[0];
    c1 += a * (float)v4[1];
    c2 += a * (float)v4[2];
    c3 += a * (float)v4[3];
  }
  h4 cv; cv[0] = (_Float16)c0; cv[1] = (_Float16)c1;
  cv[2] = (_Float16)c2; cv[3] = (_Float16)c3;
  *(h4*)(ctx_out + (size_t)b * 128 + lane * 4) = cv;
}

// ---------------------------------------------------------------------------
// Output pipeline: out = MLP([ctx@wo || src_emb])
// ---------------------------------------------------------------------------
__global__ void out_pipeline(const _Float16* __restrict__ ctx,
                             const float* __restrict__ src_emb,
                             const _Float16* __restrict__ woTg,
                             const _Float16* __restrict__ w1Tg,
                             const _Float16* __restrict__ w2Tg,
                             const float* __restrict__ b1, const float* __restrict__ b2,
                             float* __restrict__ out, int blocks_per_wg) {
  extern __shared__ _Float16 sm[];
  _Float16* woT = sm;                // 128*128
  _Float16* w1T = woT + 128 * 128;   // 256*128
  _Float16* w2T = w1T + 256 * 128;   // 128*128
  _Float16* Ac  = w2T + 128 * 128;   // 16*128
  _Float16* A2  = Ac + 16 * 128;     // 16*256
  _Float16* H   = A2 + 16 * 256;     // 16*128
  const int t = threadIdx.x, w = t >> 5;
  for (int i = t; i < 2048; i += 256) {
    lds_async_b128(woT + i * 8, woTg + i * 8);
    lds_async_b128(w2T + i * 8, w2Tg + i * 8);
  }
  for (int i = t; i < 4096; i += 256)
    lds_async_b128(w1T + i * 8, w1Tg + i * 8);
  wait_async_zero();
  __syncthreads();
  for (int blk = 0; blk < blocks_per_wg; ++blk) {
    const int r0 = (blockIdx.x * blocks_per_wg + blk) * 16;
    if (r0 >= TGN_B) break;
    for (int i = t; i < 16 * 16; i += 256) {   // ctx: h8 copies
      int row = i >> 4, c8 = (i & 15) * 8;
      *(h8*)(Ac + row * 128 + c8) = *(const h8*)(ctx + (size_t)(r0 + row) * 128 + c8);
    }
    for (int i = t; i < 16 * 32; i += 256) {   // src_emb: float4 -> h4
      int row = i >> 5, c4 = (i & 31) * 4;
      f4 v = *(const f4*)(src_emb + (size_t)(r0 + row) * 128 + c4);
      *(h4*)(A2 + row * 256 + 128 + c4) = __builtin_convertvector(v, h4);
    }
    __syncthreads();
    v8f a0 = {};
    a0 = wmma_tile(Ac, 128, woT, 128, w * 16, a0);          // ctx @ wo
    tile_to_lds_f16(a0, A2, 256, w * 16, nullptr, false);   // -> A2[:,0:128]
    __syncthreads();
    v8f a1 = {};
    a1 = wmma_tile(A2, 256, w1T, 256, w * 16, a1);
    tile_to_lds_f16(a1, H, 128, w * 16, b1, true);
    __syncthreads();
    v8f a2 = {};
    a2 = wmma_tile(H, 128, w2T, 128, w * 16, a2);
    tile_to_glb_f32(a2, out + (size_t)r0 * 128, w * 16, b2);
    __syncthreads();
  }
}

// ---------------------------------------------------------------------------
extern "C" void kernel_launch(void* const* d_in, const int* in_sizes, int n_in,
                              void* d_out, int out_size, void* d_ws, size_t ws_size,
                              hipStream_t stream) {
  const float* nf   = (const float*)d_in[0];
  const float* mem  = (const float*)d_in[1];
  const float* ef   = (const float*)d_in[2];
  const float* ts   = (const float*)d_in[3];
  const float* lu   = (const float*)d_in[4];
  const float* et   = (const float*)d_in[5];
  const int*   src  = (const int*)d_in[6];
  const int*   ngh  = (const int*)d_in[7];
  const int*   eidx = (const int*)d_in[8];
  const float* w1i  = (const float*)d_in[9];
  const float* b1i  = (const float*)d_in[10];
  const float* w2i  = (const float*)d_in[11];
  const float* b2i  = (const float*)d_in[12];
  const float* wts  = (const float*)d_in[13];
  const float* bts  = (const float*)d_in[14];
  const float* wtn  = (const float*)d_in[15];
  const float* btn  = (const float*)d_in[16];
  const float* wq   = (const float*)d_in[17];
  const float* wk   = (const float*)d_in[18];
  const float* wv   = (const float*)d_in[19];
  const float* wo   = (const float*)d_in[20];
  const float* w1o  = (const float*)d_in[21];
  const float* b1o  = (const float*)d_in[22];
  const float* w2o  = (const float*)d_in[23];
  const float* b2o  = (const float*)d_in[24];

  char* ws = (char*)d_ws;
  size_t off = 0;
  auto take = [&](size_t bytes) { char* p = ws + off; off += (bytes + 255) & ~size_t(255); return p; };
  _Float16* w1T  = (_Float16*)take(256 * 128 * 2);
  _Float16* w2T  = (_Float16*)take(128 * 128 * 2);
  _Float16* wqT  = (_Float16*)take(256 * 128 * 2);
  _Float16* wkT  = (_Float16*)take(384 * 128 * 2);
  _Float16* wvT  = (_Float16*)take(384 * 128 * 2);
  _Float16* woT  = (_Float16*)take(128 * 128 * 2);
  _Float16* w1oT = (_Float16*)take(256 * 128 * 2);
  _Float16* w2oT = (_Float16*)take(128 * 128 * 2);
  _Float16* embB = (_Float16*)take((size_t)NGH_ROWS * 128 * 2);
  _Float16* kB   = (_Float16*)take((size_t)NGH_ROWS * 128 * 2);
  _Float16* vB   = (_Float16*)take((size_t)NGH_ROWS * 128 * 2);
  float*    seB  = (float*)take((size_t)TGN_B * 128 * 4);
  float*    qB   = (float*)take((size_t)TGN_B * 128 * 4);
  _Float16* ctxB = (_Float16*)take((size_t)TGN_B * 128 * 2);

  // 1) weight conversion (f32 row-major -> f16 transposed)
  convert_wT<<<(256 * 128 + 255) / 256, 256, 0, stream>>>(w1i, w1T, 256, 128);
  convert_wT<<<(128 * 128 + 255) / 256, 256, 0, stream>>>(w2i, w2T, 128, 128);
  convert_wT<<<(256 * 128 + 255) / 256, 256, 0, stream>>>(wq, wqT, 256, 128);
  convert_wT<<<(384 * 128 + 255) / 256, 256, 0, stream>>>(wk, wkT, 384, 128);
  convert_wT<<<(384 * 128 + 255) / 256, 256, 0, stream>>>(wv, wvT, 384, 128);
  convert_wT<<<(128 * 128 + 255) / 256, 256, 0, stream>>>(wo, woT, 128, 128);
  convert_wT<<<(256 * 128 + 255) / 256, 256, 0, stream>>>(w1o, w1oT, 256, 128);
  convert_wT<<<(128 * 128 + 255) / 256, 256, 0, stream>>>(w2o, w2oT, 128, 128);

  // 2) source pipeline: 4096 rows = 32 WGs x 8 blocks of 16
  const size_t SRC_LDS = (size_t)(256*128 + 128*128 + 256*128 + 16*256 + 16*128 + 16*256) * 2;
  src_pipeline<<<32, 256, SRC_LDS, stream>>>(nf, mem, src, ts, lu, wts, bts,
                                             w1T, w2T, wqT, b1i, b2i, seB, qB, 8);

  // 3) neighbor MLP: 81920 rows = 320 WGs x 16 blocks of 16
  const size_t NGH_LDS = (size_t)(256*128 + 128*128 + 16*256 + 16*128) * 2;
  ngh_mlp<<<320, 256, NGH_LDS, stream>>>(nf, mem, ngh, w1T, w2T, b1i, b2i, embB, 16);

  // 4) k/v projection: 320 WGs x 16 blocks
  const size_t KV_LDS = (size_t)(384*128 + 384*128 + 16*384) * 2;
  kv_proj<<<320, 256, KV_LDS, stream>>>(embB, ts, et, ef, eidx, wtn, btn,
                                        wkT, wvT, kB, vB, 16);

  // 5) attention: one wave32 per batch row
  attn_kernel<<<TGN_B / 8, 256, 0, stream>>>(qB, kB, vB, ngh, ctxB);

  // 6) output pipeline
  const size_t OUT_LDS = (size_t)(128*128 + 256*128 + 128*128 + 16*128 + 16*256 + 16*128) * 2;
  out_pipeline<<<32, 256, OUT_LDS, stream>>>(ctxB, seB, woT, w1oT, w2oT,
                                             b1o, b2o, (float*)d_out, 8);

  (void)in_sizes; (void)n_in; (void)out_size; (void)ws_size;
}
